// PhysNetPretrain_4810363372626
// MI455X (gfx1250) — matrix operand order repaired and checked by reading, not compiled, not run on hardware
//
#include <hip/hip_runtime.h>
#include <hip/hip_bf16.h>

// ---------------------------------------------------------------------------
// PhysNet message-passing blocks, fused for MI455X (gfx1250).
//   - edge matmuls via v_wmma_f32_16x16x32_bf16 (f32 accumulate)
//   - one wave owns a 16-edge x 128-feature tile end-to-end (all residuals)
//   - weights pre-packed into B-fragment-major bf16 (cached broadcast loads)
//   - activations relayout C->A through a 4KB/wave LDS tile
//   - segment_sum via global_atomic_add_f32 into h_next
// ---------------------------------------------------------------------------

#define FDIM 128
#define KRBF 5
#define NBLK 5
#define NT 8          // 8 output column tiles of 16
#define KC 4          // 4 K-chunks of 32
#define PER_MAT (NT*KC*32*16)   // 16384 bf16 elements per packed 128x128 matrix
#define MATS_PER_BLK 7          // wi, wj, w1_0, w2_0, w1_1, w2_1, wd

typedef __attribute__((ext_vector_type(16))) __bf16 v16bf;
typedef __attribute__((ext_vector_type(8)))  __bf16 v8bf;
typedef __attribute__((ext_vector_type(8)))  float  v8f;

__device__ __forceinline__ float ssp(float x) {
  // shifted softplus: softplus(x) - log(2), numerically stable
  return fmaxf(x, 0.0f) + log1pf(__expf(-fabsf(x))) - 0.69314718055994531f;
}

__device__ __forceinline__ v16bf cat16(v8bf lo, v8bf hi) {
  v16bf r;
#pragma unroll
  for (int i = 0; i < 8; ++i) { r[i] = lo[i]; r[i + 8] = hi[i]; }
  return r;
}

// A fragment (16x32 bf16) from a row-major [16][128] bf16 LDS tile.
// ISA 16-bit A layout: lane = M (mod 16); half = lane>>4;
//   elems 0..7  <-> k = kc*32 + half*8 + e
//   elems 8..15 <-> k = kc*32 + 16 + half*8 + (e-8)
__device__ __forceinline__ v16bf load_a_frag(const __bf16* tile, int lane, int kc) {
  int m = lane & 15, half = lane >> 4;
  const __bf16* p = tile + m * FDIM + kc * 32 + half * 8;
  v8bf lo = *(const v8bf*)p;
  v8bf hi = *(const v8bf*)(p + 16);
  return cat16(lo, hi);
}

// B fragment (32x16 bf16) from fragment-major packed global weights.
__device__ __forceinline__ v16bf load_b_frag(const __bf16* pw, int t, int kc, int lane) {
  const __bf16* p = pw + (((t * KC + kc) * 32) + lane) * 16;
  v8bf lo = *(const v8bf*)p;
  v8bf hi = *(const v8bf*)(p + 8);
  return cat16(lo, hi);
}

__device__ __forceinline__ v8f wmma_bf16(v16bf a, v16bf b, v8f c) {
  return __builtin_amdgcn_wmma_f32_16x16x32_bf16(false, a, false, b, (short)0, c,
                                                 false, false);
}

// One ResidualLayer applied to a C-layout accumulator set:
//   m = m + (ssp(m) @ W1^T + b1) @ W2^T + b2
__device__ __forceinline__ void residual_step(v8f (&macc)[NT], __bf16* myact,
                                              const __bf16* pw1, const float* b1,
                                              const __bf16* pw2, const float* b2,
                                              int lane) {
  int col = lane & 15, rowbase = (lane >> 4) * 8;
  // stage y = ssp(m) into LDS tile (C-layout scatter -> row-major tile)
  __syncthreads();
#pragma unroll
  for (int t = 0; t < NT; ++t)
#pragma unroll
    for (int r = 0; r < 8; ++r)
      myact[(rowbase + r) * FDIM + t * 16 + col] = (__bf16)ssp(macc[t][r]);
  __syncthreads();
  v16bf afr[KC];
#pragma unroll
  for (int kc = 0; kc < KC; ++kc) afr[kc] = load_a_frag(myact, lane, kc);

  v8f t1[NT];
#pragma unroll
  for (int t = 0; t < NT; ++t) {
    float bb = b1[t * 16 + col];
    v8f acc;
#pragma unroll
    for (int r = 0; r < 8; ++r) acc[r] = bb;
#pragma unroll
    for (int kc = 0; kc < KC; ++kc)
      acc = wmma_bf16(afr[kc], load_b_frag(pw1, t, kc, lane), acc);
    t1[t] = acc;
  }
  // stage t1, second matmul accumulates onto m
  __syncthreads();
#pragma unroll
  for (int t = 0; t < NT; ++t)
#pragma unroll
    for (int r = 0; r < 8; ++r)
      myact[(rowbase + r) * FDIM + t * 16 + col] = (__bf16)t1[t][r];
  __syncthreads();
#pragma unroll
  for (int kc = 0; kc < KC; ++kc) afr[kc] = load_a_frag(myact, lane, kc);
#pragma unroll
  for (int t = 0; t < NT; ++t) {
    v8f acc = macc[t];
#pragma unroll
    for (int kc = 0; kc < KC; ++kc)
      acc = wmma_bf16(afr[kc], load_b_frag(pw2, t, kc, lane), acc);
    float bb = b2[t * 16 + col];
#pragma unroll
    for (int r = 0; r < 8; ++r) acc[r] += bb;
    macc[t] = acc;
  }
}

// ---------------------------------------------------------------------------
// Fused edge kernel: one wave per 16-edge tile.
// ---------------------------------------------------------------------------
__global__ __launch_bounds__(128) void edge_kernel(
    const int* __restrict__ esrc, const int* __restrict__ edst,
    const float* __restrict__ rbf, const float* __restrict__ k2f,
    const float* __restrict__ hi, const float* __restrict__ hj,
    const __bf16* __restrict__ pw1_0, const __bf16* __restrict__ pw2_0,
    const __bf16* __restrict__ pw1_1, const __bf16* __restrict__ pw2_1,
    const __bf16* __restrict__ pwd,
    const float* __restrict__ b1_0, const float* __restrict__ b2_0,
    const float* __restrict__ b1_1, const float* __restrict__ b2_1,
    const float* __restrict__ bd, const float* __restrict__ u,
    float* __restrict__ hout, int nTiles) {
  __shared__ __bf16 act[4][16 * FDIM];  // 4KB per wave, 16KB total
  int wave = threadIdx.x >> 5;
  int lane = threadIdx.x & 31;
  int tile = blockIdx.x * 4 + wave;
  bool valid = tile < nTiles;
  if (tile >= nTiles) tile = nTiles - 1;  // keep barriers + EXEC uniform
  int col = lane & 15, half = lane >> 4, rowbase = half * 8;
  int e0 = tile * 16;
  __bf16* myact = act[wave];

  int dstr[8], srcr[8];
#pragma unroll
  for (int r = 0; r < 8; ++r) {
    int e = e0 + rowbase + r;
    dstr[r] = edst[e];
    srcr[r] = esrc[e];
  }

  // m = hi[dst] + (rbf @ k2f^T) * hj[src]   (built directly in C layout)
  v8f macc[NT], aik[NT];
#pragma unroll
  for (int t = 0; t < NT; ++t) {
    int f = t * 16 + col;
    float kk[KRBF];
#pragma unroll
    for (int k = 0; k < KRBF; ++k) kk[k] = k2f[f * KRBF + k];
#pragma unroll
    for (int r = 0; r < 8; ++r) {
      int e = e0 + rowbase + r;
      const float* rb = rbf + (long)e * KRBF;
      float g = 0.0f;
#pragma unroll
      for (int k = 0; k < KRBF; ++k) g = fmaf(rb[k], kk[k], g);
      float av = hi[(long)dstr[r] * FDIM + f];
      float bv = hj[(long)srcr[r] * FDIM + f];
      aik[t][r] = av;
      macc[t][r] = av + g * bv;
    }
  }

  // two residual layers on m
  residual_step(macc, myact, pw1_0, b1_0, pw2_0, b2_0, lane);
  residual_step(macc, myact, pw1_1, b1_1, pw2_1, b2_1, lane);

  // new = u*ai + ssp(m) @ Wd^T + bd
  __syncthreads();
#pragma unroll
  for (int t = 0; t < NT; ++t)
#pragma unroll
    for (int r = 0; r < 8; ++r)
      myact[(rowbase + r) * FDIM + t * 16 + col] = (__bf16)ssp(macc[t][r]);
  __syncthreads();
  v16bf afr[KC];
#pragma unroll
  for (int kc = 0; kc < KC; ++kc) afr[kc] = load_a_frag(myact, lane, kc);

  v8f nacc[NT];
#pragma unroll
  for (int t = 0; t < NT; ++t) {
    int f = t * 16 + col;
    float uf = u[f], bdf = bd[f];
    v8f acc;
#pragma unroll
    for (int r = 0; r < 8; ++r) acc[r] = uf * aik[t][r] + bdf;
#pragma unroll
    for (int kc = 0; kc < KC; ++kc)
      acc = wmma_bf16(afr[kc], load_b_frag(pwd, t, kc, lane), acc);
    nacc[t] = acc;
  }

  // same residual layers reused on new (as in reference)
  residual_step(nacc, myact, pw1_0, b1_0, pw2_0, b2_0, lane);
  residual_step(nacc, myact, pw1_1, b1_1, pw2_1, b2_1, lane);

  // segment_sum: scatter-add into h_next
  if (valid) {
#pragma unroll
    for (int t = 0; t < NT; ++t)
#pragma unroll
      for (int r = 0; r < 8; ++r)
        __hip_atomic_fetch_add(&hout[(long)dstr[r] * FDIM + t * 16 + col],
                               nacc[t][r], __ATOMIC_RELAXED,
                               __HIP_MEMORY_SCOPE_AGENT);
  }
}

// ---------------------------------------------------------------------------
// Node kernel: x = ssp(h); hi = x@Wi^T+bi; hj = x@Wj^T+bj (WMMA, one wave/tile)
// ---------------------------------------------------------------------------
__global__ __launch_bounds__(128) void node_kernel(
    const float* __restrict__ h, const __bf16* __restrict__ pwi,
    const float* __restrict__ bi, const __bf16* __restrict__ pwj,
    const float* __restrict__ bj, float* __restrict__ hi,
    float* __restrict__ hj, int nTiles) {
  __shared__ __bf16 act[4][16 * FDIM];
  int wave = threadIdx.x >> 5;
  int lane = threadIdx.x & 31;
  int tile = blockIdx.x * 4 + wave;
  if (tile >= nTiles) tile = nTiles - 1;  // benign duplicate (same values)
  int col = lane & 15, rowbase = (lane >> 4) * 8;
  int n0 = tile * 16;
  __bf16* myact = act[wave];

  __syncthreads();
#pragma unroll
  for (int r = 0; r < 16; ++r) {
    const float* src = h + ((long)(n0 + r)) * FDIM + lane * 4;
    float4 v = *(const float4*)src;
    __bf16* d = myact + r * FDIM + lane * 4;
    d[0] = (__bf16)ssp(v.x);
    d[1] = (__bf16)ssp(v.y);
    d[2] = (__bf16)ssp(v.z);
    d[3] = (__bf16)ssp(v.w);
  }
  __syncthreads();
  v16bf afr[KC];
#pragma unroll
  for (int kc = 0; kc < KC; ++kc) afr[kc] = load_a_frag(myact, lane, kc);

#pragma unroll
  for (int t = 0; t < NT; ++t) {
    int f = t * 16 + col;
    float bbi = bi[f], bbj = bj[f];
    v8f ai, aj;
#pragma unroll
    for (int r = 0; r < 8; ++r) { ai[r] = bbi; aj[r] = bbj; }
#pragma unroll
    for (int kc = 0; kc < KC; ++kc) {
      ai = wmma_bf16(afr[kc], load_b_frag(pwi, t, kc, lane), ai);
      aj = wmma_bf16(afr[kc], load_b_frag(pwj, t, kc, lane), aj);
    }
#pragma unroll
    for (int r = 0; r < 8; ++r) {
      long o = ((long)(n0 + rowbase + r)) * FDIM + f;
      hi[o] = ai[r];
      hj[o] = aj[r];
    }
  }
}

// ---------------------------------------------------------------------------
// One-time prep kernels
// ---------------------------------------------------------------------------
__global__ void prep_weights(const float* __restrict__ wi,
                             const float* __restrict__ wj,
                             const float* __restrict__ rw1,
                             const float* __restrict__ rw2,
                             const float* __restrict__ wd,
                             __bf16* __restrict__ packed) {
  int idx = blockIdx.x * blockDim.x + threadIdx.x;
  if (idx >= NBLK * MATS_PER_BLK * PER_MAT) return;
  int b = idx / (MATS_PER_BLK * PER_MAT);
  int rem = idx % (MATS_PER_BLK * PER_MAT);
  int mat = rem / PER_MAT;
  int pe = rem % PER_MAT;
  int e = pe & 15, lane = (pe >> 4) & 31, kc = (pe >> 9) & 3, t = pe >> 11;
  int n = t * 16 + (lane & 15);
  int k = kc * 32 + (lane >> 4) * 8 + (e & 7) + (e >> 3) * 16;  // B layout: lanes 0-15 K=0..7,16..23; 16-31 K=8..15,24..31
  const float* src;
  switch (mat) {
    case 0: src = wi + b * FDIM * FDIM; break;
    case 1: src = wj + b * FDIM * FDIM; break;
    case 2: src = rw1 + (b * 2 + 0) * FDIM * FDIM; break;
    case 3: src = rw2 + (b * 2 + 0) * FDIM * FDIM; break;
    case 4: src = rw1 + (b * 2 + 1) * FDIM * FDIM; break;
    case 5: src = rw2 + (b * 2 + 1) * FDIM * FDIM; break;
    default: src = wd + b * FDIM * FDIM; break;
  }
  packed[idx] = (__bf16)src[n * FDIM + k];  // B[k][n] = W[n][k] for x@W^T
}

__global__ void rbf_kernel(const int* __restrict__ esrc,
                           const int* __restrict__ edst,
                           const float* __restrict__ pos,
                           float* __restrict__ rbf, int E) {
  int e = blockIdx.x * blockDim.x + threadIdx.x;
  if (e >= E) return;
  int s = esrc[e], d = edst[e];
  float dx = pos[s * 3 + 0] - pos[d * 3 + 0];
  float dy = pos[s * 3 + 1] - pos[d * 3 + 1];
  float dz = pos[s * 3 + 2] - pos[d * 3 + 2];
  float dist = sqrtf(dx * dx + dy * dy + dz * dz);
  float x = dist * 0.1f;
  float x3 = x * x * x, x4 = x3 * x, x5 = x4 * x;
  float cut = (x < 1.0f) ? (1.0f - 6.0f * x5 + 15.0f * x4 - 10.0f * x3) : 0.0f;
  float emd = __expf(-dist);
  float emc = __expf(-10.0f);
  float width = 2.5f / (1.0f - emc);
  width = width * width;
  float step = (emc - 1.0f) * 0.25f;
#pragma unroll
  for (int k = 0; k < KRBF; ++k) {
    float c = 1.0f + step * (float)k;
    float dlt = emd - c;
    rbf[(long)e * KRBF + k] = cut * __expf(-width * dlt * dlt);
  }
}

__global__ void gather_emb(const int* __restrict__ at,
                           const float* __restrict__ emb,
                           float* __restrict__ h, int total) {
  int i = blockIdx.x * blockDim.x + threadIdx.x;
  if (i >= total) return;
  int n = i >> 7, f = i & (FDIM - 1);
  h[i] = emb[at[n] * FDIM + f];
}

__global__ void zero_kernel(float* __restrict__ p, int total) {
  int i = blockIdx.x * blockDim.x + threadIdx.x;
  if (i < total) p[i] = 0.0f;
}

// ---------------------------------------------------------------------------
extern "C" void kernel_launch(void* const* d_in, const int* in_sizes, int n_in,
                              void* d_out, int out_size, void* d_ws, size_t ws_size,
                              hipStream_t stream) {
  const int*   atom_types = (const int*)d_in[0];
  const int*   esrc       = (const int*)d_in[1];
  const int*   edst       = (const int*)d_in[2];
  const float* pos        = (const float*)d_in[3];
  const float* emb        = (const float*)d_in[4];
  const float* k2f        = (const float*)d_in[5];   // [5][128][5]
  const float* wi         = (const float*)d_in[6];
  const float* bi         = (const float*)d_in[7];
  const float* wj         = (const float*)d_in[8];
  const float* bj         = (const float*)d_in[9];
  const float* wd         = (const float*)d_in[10];
  const float* bd         = (const float*)d_in[11];
  const float* u          = (const float*)d_in[12];
  const float* rw1        = (const float*)d_in[13];  // [5][2][128][128]
  const float* rb1        = (const float*)d_in[14];
  const float* rw2        = (const float*)d_in[15];
  const float* rb2        = (const float*)d_in[16];
  float* out = (float*)d_out;

  const int N = in_sizes[0];
  const int E = in_sizes[1];
  const int NF = N * FDIM;

  // workspace carve (~120 MB): rbf | hi | hj | hA | hB | packed weights
  char* ws = (char*)d_ws;
  float* rbf = (float*)ws;            ws += (size_t)E * KRBF * sizeof(float);
  float* hib = (float*)ws;            ws += (size_t)NF * sizeof(float);
  float* hjb = (float*)ws;            ws += (size_t)NF * sizeof(float);
  float* hA  = (float*)ws;            ws += (size_t)NF * sizeof(float);
  float* hB  = (float*)ws;            ws += (size_t)NF * sizeof(float);
  __bf16* packed = (__bf16*)ws;

  {
    int tot = NBLK * MATS_PER_BLK * PER_MAT;
    prep_weights<<<(tot + 255) / 256, 256, 0, stream>>>(wi, wj, rw1, rw2, wd, packed);
  }
  rbf_kernel<<<(E + 255) / 256, 256, 0, stream>>>(esrc, edst, pos, rbf, E);
  gather_emb<<<(NF + 255) / 256, 256, 0, stream>>>(atom_types, emb, hA, NF);

  int nodeTiles = (N + 15) / 16;
  int edgeTiles = (E + 15) / 16;
  int nodeGrid = (nodeTiles + 3) / 4;
  int edgeGrid = (edgeTiles + 3) / 4;

  float* hcur = hA;
  float* hnxt = hB;
  for (int b = 0; b < NBLK; ++b) {
    const __bf16* pb = packed + (size_t)(b * MATS_PER_BLK) * PER_MAT;
    node_kernel<<<nodeGrid, 128, 0, stream>>>(
        hcur, pb + 0 * PER_MAT, bi + b * FDIM, pb + 1 * PER_MAT, bj + b * FDIM,
        hib, hjb, nodeTiles);

    float* tgt = (b == NBLK - 1) ? out : hnxt;
    zero_kernel<<<(NF + 255) / 256, 256, 0, stream>>>(tgt, NF);
    edge_kernel<<<edgeGrid, 128, 0, stream>>>(
        esrc, edst, rbf, k2f + b * FDIM * KRBF, hib, hjb,
        pb + 2 * PER_MAT, pb + 3 * PER_MAT, pb + 4 * PER_MAT, pb + 5 * PER_MAT,
        pb + 6 * PER_MAT,
        rb1 + (b * 2 + 0) * FDIM, rb2 + (b * 2 + 0) * FDIM,
        rb1 + (b * 2 + 1) * FDIM, rb2 + (b * 2 + 1) * FDIM,
        bd + b * FDIM, u + b * FDIM, tgt, edgeTiles);

    float* old = hcur;
    hcur = tgt;
    hnxt = old;
  }
  (void)n_in; (void)out_size; (void)ws_size;
}
